// DeMBR_28484223107660
// MI455X (gfx1250) — matrix operand order repaired and checked by reading, compiled, not run on hardware
//
#include <hip/hip_runtime.h>

typedef __attribute__((ext_vector_type(16))) __bf16 v16bf;
typedef __attribute__((ext_vector_type(8)))  __bf16 v8bf;
typedef __attribute__((ext_vector_type(4)))  __bf16 v4bf;
typedef __attribute__((ext_vector_type(8)))  float  v8f;

#define NN 4096
#define DD 64
#define BM 32
#define BK 32
#define APAD 40         // bf16 row stride for LDS tiles (b128-aligned frag loads)
#define NITER (NN / BK)

struct Combo {
  const float*  R;     // [N,N] behavior matrix (fp32, streamed)
  const __bf16* Xhi;   // [D,N] transposed hi-part of dense operand
  const __bf16* Xlo;   // [D,N] transposed lo-part
  const float*  prev;  // layer-1 result (layer 2 only)
  const float*  addv;  // virtual-branch add (layer-2 user pass only)
  float* deg;          // degree (+EPS) vector in workspace
  float* out;          // [N,D] destination
};
struct PassArgs { Combo c[7]; };
struct PrepArgs { const float* src[16]; __bf16* hi[16]; __bf16* lo[16]; };

__device__ __forceinline__ v8f wmma_bf16(v16bf a, v16bf b, v8f c) {
  return __builtin_amdgcn_wmma_f32_16x16x32_bf16(
      false, a, false, b, (short)0, c, false, false);
}

// CDNA5 async DMA: global -> LDS, 16B per lane, tracked by ASYNCcnt.
// vdst = LDS byte address VGPR (low 32 bits of flat shared pointer),
// vaddr = 64-bit global address, saddr = off.   (ISA 08_async_tensor §4)
__device__ __forceinline__ void async_g2l_b128(unsigned lds_off, const void* g) {
  asm volatile("global_load_async_to_lds_b128 %0, %1, off"
               :: "v"(lds_off), "v"((unsigned long long)(size_t)g)
               : "memory");
}
__device__ __forceinline__ void wait_async0() {
#if __has_builtin(__builtin_amdgcn_s_wait_asynccnt)
  __builtin_amdgcn_s_wait_asynccnt(0);
#else
  asm volatile("s_wait_asynccnt 0x0" ::: "memory");
#endif
}

// Convert X[N,D] fp32 -> Xt_hi/Xt_lo [D,N] bf16 (fragment-friendly, k-contiguous).
__global__ __launch_bounds__(256) void xt_prep(PrepArgs pp) {
  const float*  src = pp.src[blockIdx.y];
  __bf16* hi = pp.hi[blockIdx.y];
  __bf16* lo = pp.lo[blockIdx.y];
  const int t  = threadIdx.x;
  const int n  = t & 63;
  const int k0 = blockIdx.x * 64 + (t >> 6) * 16;
  v8bf h0, h1, l0, l1;
#pragma unroll
  for (int j = 0; j < 8; ++j) {
    const float f = src[(size_t)(k0 + j) * DD + n];
    const __bf16 h = (__bf16)f;
    h0[j] = h; l0[j] = (__bf16)(f - (float)h);
  }
#pragma unroll
  for (int j = 0; j < 8; ++j) {
    const float f = src[(size_t)(k0 + 8 + j) * DD + n];
    const __bf16 h = (__bf16)f;
    h1[j] = h; l1[j] = (__bf16)(f - (float)h);
  }
  const size_t o = (size_t)n * NN + k0;
  *(v8bf*)(hi + o) = h0; *(v8bf*)(hi + o + 8) = h1;
  *(v8bf*)(lo + o) = l0; *(v8bf*)(lo + o + 8) = l1;
}

template <int TRANS, int LAY2>
__global__ __launch_bounds__(256) void dembr_pass(PassArgs pa) {
  const Combo cb = pa.c[blockIdx.y];
  const int m0 = blockIdx.x * BM;

  __shared__ __bf16 sAhi[2][BM][APAD];
  __shared__ __bf16 sAlo[2][BM][APAD];
  __shared__ __bf16 sBhi[2][DD][APAD];
  __shared__ __bf16 sBlo[2][DD][APAD];
  __shared__ float  sdeg[BM];

  const int t    = threadIdx.x;
  const int lane = t & 31;
  const int wv   = t >> 5;
  const int mt   = (wv >> 2) << 4;   // 0 or 16
  const int nt   = (wv & 3) << 4;    // 0,16,32,48
  const int hf   = lane >> 4;
  const int l16  = lane & 15;

  if (t < BM) sdeg[t] = LAY2 ? cb.deg[m0 + t] : 0.0f;

  float rowpart = 0.0f;
  float colpart[4] = {0.f, 0.f, 0.f, 0.f};
  v8f acc = {};

  // ---- helpers -------------------------------------------------------------
  auto loadA = [&](int k0) -> float4 {   // fetch one R chunk into VGPRs
    if (TRANS == 0)
      return *(const float4*)(cb.R + (size_t)(m0 + (t >> 3)) * NN +
                              (k0 + ((t & 7) << 2)));
    else
      return *(const float4*)(cb.R + (size_t)(k0 + (t >> 3)) * NN +
                              (m0 + ((t & 7) << 2)));
  };
  auto stageA = [&](const float4& v, int bsel) {  // split fp32 -> bf16 hi/lo
    if (TRANS == 0) {
      const int m = t >> 3, kk = (t & 7) << 2;
      const __bf16 hx = (__bf16)v.x, hy = (__bf16)v.y,
                   hz = (__bf16)v.z, hw = (__bf16)v.w;
      const v4bf hv = {hx, hy, hz, hw};
      const v4bf lv = {(__bf16)(v.x - (float)hx), (__bf16)(v.y - (float)hy),
                       (__bf16)(v.z - (float)hz), (__bf16)(v.w - (float)hw)};
      *(v4bf*)&sAhi[bsel][m][kk] = hv;
      *(v4bf*)&sAlo[bsel][m][kk] = lv;
      if (LAY2 == 0) rowpart += (v.x + v.y) + (v.z + v.w);
    } else {
      const int kk = t >> 3, m = (t & 7) << 2;
      const float f[4] = {v.x, v.y, v.z, v.w};
#pragma unroll
      for (int j = 0; j < 4; ++j) {
        const __bf16 h = (__bf16)f[j];
        sAhi[bsel][m + j][kk] = h;
        sAlo[bsel][m + j][kk] = (__bf16)(f[j] - (float)h);
        if (LAY2 == 0) colpart[j] += f[j];
      }
    }
  };
  auto stageB = [&](int k0, int bsel) {  // async DMA of pre-split operand
    const int n    = t >> 2;
    const int koff = (t & 3) << 3;
    const size_t o = (size_t)n * NN + k0 + koff;
    async_g2l_b128((unsigned)(size_t)&sBhi[bsel][n][koff], cb.Xhi + o);
    async_g2l_b128((unsigned)(size_t)&sBlo[bsel][n][koff], cb.Xlo + o);
  };

  // ---- prologue: fill buffer 0 --------------------------------------------
  {
    stageB(0, 0);
    const float4 v0 = loadA(0);
    stageA(v0, 0);
    wait_async0();
  }
  __syncthreads();

  // ---- main loop: 1 barrier/iter, async B + pipelined A into buf^1 --------
  int buf = 0;
  for (int i = 0; i < NITER; ++i) {
    const int k0 = i * BK;
    const bool more = (i + 1 < NITER);
    float4 vnext{};
    if (more) {
      stageB(k0 + BK, buf ^ 1);          // DMA next B chunk (ASYNCcnt)
      vnext = loadA(k0 + BK);            // next R chunk -> VGPRs
      const int kp = (k0 + 2 * BK < NN) ? (k0 + 2 * BK) : k0;
      if (TRANS == 0)
        __builtin_prefetch(cb.R + (size_t)(m0 + (t >> 3)) * NN +
                           (kp + ((t & 7) << 2)), 0, 3);
      else
        __builtin_prefetch(cb.R + (size_t)(kp + (t >> 3)) * NN +
                           (m0 + ((t & 7) << 2)), 0, 3);
    }

    // fragments: aligned ds_load_b128 pairs (ISA 7.12.2 A/B layouts)
    const int mA = mt + l16;
    const int nB = nt + l16;
    const v8bf a0 = *(const v8bf*)&sAhi[buf][mA][hf * 8];
    const v8bf a1 = *(const v8bf*)&sAhi[buf][mA][16 + hf * 8];
    const v8bf e0 = *(const v8bf*)&sAlo[buf][mA][hf * 8];
    const v8bf e1 = *(const v8bf*)&sAlo[buf][mA][16 + hf * 8];
    const v8bf b0 = *(const v8bf*)&sBhi[buf][nB][hf * 8];
    const v8bf b1 = *(const v8bf*)&sBhi[buf][nB][16 + hf * 8];
    const v8bf d0 = *(const v8bf*)&sBlo[buf][nB][hf * 8];
    const v8bf d1 = *(const v8bf*)&sBlo[buf][nB][16 + hf * 8];
    const v16bf ahi = __builtin_shufflevector(a0, a1, 0,1,2,3,4,5,6,7,8,9,10,11,12,13,14,15);
    const v16bf alo = __builtin_shufflevector(e0, e1, 0,1,2,3,4,5,6,7,8,9,10,11,12,13,14,15);
    const v16bf bhi = __builtin_shufflevector(b0, b1, 0,1,2,3,4,5,6,7,8,9,10,11,12,13,14,15);
    const v16bf blo = __builtin_shufflevector(d0, d1, 0,1,2,3,4,5,6,7,8,9,10,11,12,13,14,15);

    acc = wmma_bf16(ahi, bhi, acc);   // hi*hi
    acc = wmma_bf16(ahi, blo, acc);   // hi*lo
    acc = wmma_bf16(alo, bhi, acc);   // lo*hi  -> fp32-class accuracy

    if (more) {
      stageA(vnext, buf ^ 1);          // convert + ds_store into other buffer
      wait_async0();                   // B DMA for buf^1 complete
    }
    __syncthreads();                   // publish buf^1, retire reads of buf
    buf ^= 1;
  }

  // ---- layer 1: reduce degree partials, add EPS, publish -------------------
  if (LAY2 == 0) {
    if (TRANS == 0) {
      atomicAdd(&sdeg[t >> 3], rowpart);
    } else {
      const int m = (t & 7) << 2;
      atomicAdd(&sdeg[m + 0], colpart[0]);
      atomicAdd(&sdeg[m + 1], colpart[1]);
      atomicAdd(&sdeg[m + 2], colpart[2]);
      atomicAdd(&sdeg[m + 3], colpart[3]);
    }
    __syncthreads();
    if (t < BM) {
      const float s = sdeg[t] + 1e-8f;
      sdeg[t] = s;
      cb.deg[m0 + t] = s;
    }
    __syncthreads();
  }

  // ---- epilogue: 1/deg scale; layer-2 fuses (prev+cur)/2 (+add) ------------
  const bool hasadd = (LAY2 != 0) && (cb.addv != nullptr);
#pragma unroll
  for (int r = 0; r < 8; ++r) {
    const int m = mt + (hf << 3) + r;
    const int n = nt + l16;
    float v = acc[r] / sdeg[m];
    const size_t idx = (size_t)(m0 + m) * DD + n;
    if (LAY2) {
      v = (cb.prev[idx] + v) * 0.5f;
      if (hasadd) v += cb.addv[idx];
    }
    cb.out[idx] = v;
  }
}

extern "C" void kernel_launch(void* const* d_in, const int* in_sizes, int n_in,
                              void* d_out, int out_size, void* d_ws, size_t ws_size,
                              hipStream_t stream) {
  (void)in_sizes; (void)n_in; (void)out_size; (void)ws_size;
  const float* user_emb = (const float*)d_in[0];
  const float* item_emb = (const float*)d_in[1];
  const float* R[7];
  for (int b = 0; b < 7; ++b) R[b] = (const float*)d_in[2 + b];
  const float* addv3[3] = {(const float*)d_in[9], (const float*)d_in[10],
                           (const float*)d_in[11]};
  float* out = (float*)d_out;
  float* w   = (float*)d_ws;

  const size_t ND = (size_t)NN * DD;
  // fp32 workspace: u1[7] | i1[7] | deg_u[7] | deg_i[7]
  float* u1[7]; float* i1[7]; float* dgu[7]; float* dgi[7];
  for (int b = 0; b < 7; ++b) {
    u1[b]  = w + (size_t)b * ND;
    i1[b]  = w + (size_t)(7 + b) * ND;
    dgu[b] = w + 14 * ND + (size_t)b * NN;
    dgi[b] = w + 14 * ND + (size_t)(7 + b) * NN;
  }
  // bf16 workspace: 16 operand slots of transposed hi/lo [64][4096]
  __bf16* bfbase = (__bf16*)(w + 14 * ND + 14 * NN);
  const size_t XSZ = (size_t)DD * NN;
  auto Xhi = [&](int s) { return bfbase + (size_t)s * 2 * XSZ; };
  auto Xlo = [&](int s) { return bfbase + (size_t)s * 2 * XSZ + XSZ; };
  // slots: 0=user_emb, 1=item_emb, 2+b=u1[b], 9+b=i1[b]

  PrepArgs P1{};
  P1.src[0] = user_emb; P1.hi[0] = Xhi(0); P1.lo[0] = Xlo(0);
  P1.src[1] = item_emb; P1.hi[1] = Xhi(1); P1.lo[1] = Xlo(1);
  xt_prep<<<dim3(NN / 64, 2), 256, 0, stream>>>(P1);

  PassArgs L1r{}, L1c{};
  for (int b = 0; b < 7; ++b) {
    L1r.c[b] = { R[b], Xhi(1), Xlo(1), nullptr, nullptr, dgu[b], u1[b] };
    L1c.c[b] = { R[b], Xhi(0), Xlo(0), nullptr, nullptr, dgi[b], i1[b] };
  }
  dim3 grid(NN / BM, 7), block(256);
  dembr_pass<0, 0><<<grid, block, 0, stream>>>(L1r);
  dembr_pass<1, 0><<<grid, block, 0, stream>>>(L1c);

  PrepArgs P2{};
  for (int b = 0; b < 7; ++b) {
    P2.src[b]     = u1[b]; P2.hi[b]     = Xhi(2 + b); P2.lo[b]     = Xlo(2 + b);
    P2.src[7 + b] = i1[b]; P2.hi[7 + b] = Xhi(9 + b); P2.lo[7 + b] = Xlo(9 + b);
  }
  xt_prep<<<dim3(NN / 64, 14), 256, 0, stream>>>(P2);

  PassArgs L2r{}, L2c{};
  for (int b = 0; b < 7; ++b) {
    float* ou = (b < 4) ? out + (size_t)b * ND
                        : out + (size_t)(8 + (b - 4)) * ND;
    float* oi = (b < 4) ? out + (size_t)(4 + b) * ND
                        : out + (size_t)(11 + (b - 4)) * ND;
    const float* av = (b >= 4) ? addv3[b - 4] : nullptr;
    L2r.c[b] = { R[b], Xhi(9 + b), Xlo(9 + b), u1[b], av,      dgu[b], ou };
    L2c.c[b] = { R[b], Xhi(2 + b), Xlo(2 + b), i1[b], nullptr, dgi[b], oi };
  }
  dembr_pass<0, 1><<<grid, block, 0, stream>>>(L2r);
  dembr_pass<1, 1><<<grid, block, 0, stream>>>(L2c);
}
